// T5Attention_1932735283623
// MI455X (gfx1250) — matrix operand order repaired and checked by reading, not compiled, not run on hardware
//
#include <hip/hip_runtime.h>
#include <hip/hip_bf16.h>

// ---------------------------------------------------------------------------
// CDNA5 (gfx1250) T5 encoder self-attention, bf16 WMMA path, wave32.
// ---------------------------------------------------------------------------

typedef __attribute__((ext_vector_type(16))) __bf16       v16bf;
typedef __attribute__((ext_vector_type(8)))  float        v8f;
typedef __attribute__((ext_vector_type(4)))  unsigned int u32x4;
typedef __attribute__((ext_vector_type(2)))  unsigned int u32x2;

// TDM descriptor group vector types
typedef __attribute__((ext_vector_type(4))) unsigned int tdm_u4;
typedef __attribute__((ext_vector_type(8))) int          tdm_i8;
typedef __attribute__((ext_vector_type(4))) int          tdm_i4;

#if defined(__has_builtin)
#if __has_builtin(__builtin_amdgcn_tensor_load_to_lds) && __has_builtin(__builtin_amdgcn_s_wait_tensorcnt)
#define HAVE_TDM 1
#endif
#endif
#ifndef HAVE_TDM
#define HAVE_TDM 0
#endif

union FragU { u32x4 u[2]; v16bf v; };

__device__ __forceinline__ v8f wmma_bf16(v16bf a, v16bf b, v8f c) {
  // (neg_a, A, neg_b, B, c_mod, C, reuse_a, reuse_b)
  return __builtin_amdgcn_wmma_f32_16x16x32_bf16(false, a, false, b, (short)0, c,
                                                 false, false);
}

// All-ones 32x16 B fragment (bf16 1.0 = 0x3F80) — used for WMMA row-sums.
__device__ __forceinline__ v16bf ones_frag() {
  FragU f;
  #pragma unroll
  for (int j = 0; j < 4; ++j) { f.u[0][j] = 0x3F803F80u; f.u[1][j] = 0x3F803F80u; }
  return f.v;
}

// A fragment 16x32 (MxK), row-major source with leading dim ld (elements).
// ISA 7.12.2: lane m=lane&15 ; lanes<16 hold K {0..7,16..23}, lanes>=16 hold {8..15,24..31}.
__device__ __forceinline__ v16bf load_frag_a(const __bf16* base, int ld) {
  const int lane = threadIdx.x & 31;
  const int m    = lane & 15;
  const int hi   = lane >> 4;
  const __bf16* p = base + (size_t)m * ld + hi * 8;
  FragU f;
  f.u[0] = *(const u32x4*)(p);        // K = hi*8 .. hi*8+7
  f.u[1] = *(const u32x4*)(p + 16);   // K = 16+hi*8 ..
  return f.v;
}

// B fragment 32x16 (KxN) sourced from B^T storage: row-major [N][K], leading dim ld.
// lanes<16: n=lane, K {0..7,16..23} ; lanes>=16: n=lane-16, K {8..15,24..31}.
__device__ __forceinline__ v16bf load_frag_b_nt(const __bf16* base, int ld) {
  const int lane = threadIdx.x & 31;
  const int n    = lane & 15;
  const int hi   = lane >> 4;
  const __bf16* p = base + (size_t)n * ld + hi * 8;
  FragU f;
  f.u[0] = *(const u32x4*)(p);
  f.u[1] = *(const u32x4*)(p + 16);
  return f.v;
}

#if HAVE_TDM
// Issue a 2D TDM tile load Global -> LDS.  data_size = 2B (bf16).
// LDS-side padding: pad_interval = 32 DWORDs (one 64-elem row = 128B),
// pad_amount = 4 DWORDs (8 bf16)  =>  effective LDS row stride = 72 elems.
__device__ __forceinline__ void tdm_load_2d_bf16(unsigned lds_off,
                                                 unsigned long long gaddr,
                                                 unsigned tensor_d0, unsigned tensor_d1,
                                                 unsigned long long d0_stride,
                                                 unsigned tile_d0, unsigned tile_d1) {
  // ----- D# group 0 (128b): count=1 | lds_addr | global_addr | type=2 -----
  unsigned __int128 g0 = 0;
  g0 |= (unsigned __int128)1u;                                  // count = 1
  g0 |= (unsigned __int128)lds_off << 32;                       // lds_addr
  g0 |= (unsigned __int128)(gaddr & ((1ull << 57) - 1)) << 64;  // global_addr
  g0 |= (unsigned __int128)2u << 126;                           // type = image
  // ----- D# group 1 (256b) -----
  const unsigned flags = (1u << 16)   // data_size = 2 bytes
                       | (1u << 20)   // pad_enable
                       | (4u << 22)   // pad_interval: code 4 -> 32 DWORDs
                       | (3u << 25);  // pad_amount:   code 3 -> 4 DWORDs
  unsigned __int128 g1lo = 0, g1hi = 0;
  g1lo |= (unsigned __int128)flags;
  g1lo |= (unsigned __int128)tensor_d0 << 48;                   // tensor_dim0
  g1lo |= (unsigned __int128)tensor_d1 << 80;                   // tensor_dim1
  g1lo |= (unsigned __int128)(tile_d0 & 0xFFFFu) << 112;        // tile_dim0
  g1hi |= (unsigned __int128)(tile_d1 & 0xFFFFu);               // tile_dim1
  g1hi |= (unsigned __int128)(d0_stride & ((1ull << 48) - 1)) << 32; // dim0 stride

  tdm_u4 v0;
  tdm_i8 v1;
  tdm_i4 v2 = {0, 0, 0, 0};
  tdm_i4 v3 = {0, 0, 0, 0};
  __builtin_memcpy(&v0, &g0, 16);
  unsigned __int128 g1[2] = { g1lo, g1hi };
  __builtin_memcpy(&v1, g1, 32);
#if __clang_major__ >= 23
  tdm_i8 v4 = {0, 0, 0, 0, 0, 0, 0, 0};
  __builtin_amdgcn_tensor_load_to_lds(v0, v1, v2, v3, v4, 0);
#else
  __builtin_amdgcn_tensor_load_to_lds(v0, v1, v2, v3, 0);
#endif
}
#endif  // HAVE_TDM

// ---------------------------------------------------------------------------
// fp32 -> bf16 conversion, 4 elements / thread (b128 global loads)
// ---------------------------------------------------------------------------
__global__ void cvt_f32_bf16_x4(const float4* __restrict__ in,
                                u32x2* __restrict__ out, int n4) {
  int i = blockIdx.x * blockDim.x + threadIdx.x;
  if (i >= n4) return;
  float4 f = in[i];
  union { u32x2 u; __bf16 e[4]; } o;
  o.e[0] = (__bf16)f.x; o.e[1] = (__bf16)f.y;
  o.e[2] = (__bf16)f.z; o.e[3] = (__bf16)f.w;
  out[i] = o.u;
}

// ---------------------------------------------------------------------------
// NT-GEMM: C[M,N] = A[M,K] * B[N,K]^T   (A,B bf16 row-major; C = OutT row-major)
// Block tile 128x64, BK=32, 256 threads = 8 waves in 4x2, wave tile 32x32.
// ---------------------------------------------------------------------------
template <typename OutT>
__global__ __launch_bounds__(256) void gemm_nt_bf16(const __bf16* __restrict__ A,
                                                    const __bf16* __restrict__ Bm,
                                                    OutT* __restrict__ C,
                                                    int M, int N, int K) {
  constexpr int BM = 128, BN = 64, BK = 32;
  constexpr int LDA = BK + 8;   // 40 elems = 80B, multiple of 16B
  constexpr int LDB = BK + 8;
  __shared__ __align__(16) __bf16 As[BM * LDA];
  __shared__ __align__(16) __bf16 Bs[BN * LDB];

  const int bm   = blockIdx.y * BM;
  const int bn   = blockIdx.x * BN;
  const int t    = threadIdx.x;
  const int wave = t >> 5;
  const int wm   = wave >> 1;      // 0..3
  const int wn   = wave & 1;       // 0..1
  const int lane = t & 31;
  const int colL = lane & 15;
  const int rowHi = (lane >> 4) * 8;

  v8f acc[2][2] = {};

  for (int k0 = 0; k0 < K; k0 += BK) {
    { // A tile: 128x32, 16 bf16 per thread
      const int row = t >> 1, half = t & 1;
      const u32x4* g = (const u32x4*)(A + (size_t)(bm + row) * K + k0 + half * 16);
      u32x4* s = (u32x4*)(As + row * LDA + half * 16);
      s[0] = g[0];
      s[1] = g[1];
      if (k0 + BK < K)  // global_prefetch_b8 for the next K-slab
        __builtin_prefetch(A + (size_t)(bm + row) * K + k0 + BK, 0, 0);
    }
    { // B tile: 64x32, 8 bf16 per thread
      const int row = t >> 2, q = t & 3;
      const u32x4* g = (const u32x4*)(Bm + (size_t)(bn + row) * K + k0 + q * 8);
      u32x4* s = (u32x4*)(Bs + row * LDB + q * 8);
      s[0] = g[0];
      if (k0 + BK < K)
        __builtin_prefetch(Bm + (size_t)(bn + row) * K + k0 + BK, 0, 0);
    }
    __syncthreads();

    v16bf a0 = load_frag_a(As + (wm * 32 + 0)  * LDA, LDA);
    v16bf a1 = load_frag_a(As + (wm * 32 + 16) * LDA, LDA);
    v16bf b0 = load_frag_b_nt(Bs + (wn * 32 + 0)  * LDB, LDB);
    v16bf b1 = load_frag_b_nt(Bs + (wn * 32 + 16) * LDB, LDB);

    acc[0][0] = wmma_bf16(a0, b0, acc[0][0]);
    acc[0][1] = wmma_bf16(a0, b1, acc[0][1]);
    acc[1][0] = wmma_bf16(a1, b0, acc[1][0]);
    acc[1][1] = wmma_bf16(a1, b1, acc[1][1]);

    __syncthreads();
  }

  // Epilogue: C layout — vgpr r, lanes<16 -> row r, lanes>=16 -> row r+8; col = lane&15.
  #pragma unroll
  for (int ti = 0; ti < 2; ++ti)
    #pragma unroll
    for (int tj = 0; tj < 2; ++tj)
      #pragma unroll
      for (int r = 0; r < 8; ++r) {
        const int gr = bm + wm * 32 + ti * 16 + rowHi + r;
        const int gc = bn + wn * 32 + tj * 16 + colL;
        C[(size_t)gr * N + gc] = static_cast<OutT>(acc[ti][tj][r]);
      }
}

// ---------------------------------------------------------------------------
// T5 relative-position bucket (bidirectional, 32 buckets, max_distance 128)
// Saturates for |rel| >= 91, so a clamped [-128,127] LUT is exact.
// ---------------------------------------------------------------------------
__device__ __forceinline__ int rel_bucket(int rel) {
  int ret = (rel > 0) ? 16 : 0;
  int ap  = (rel < 0) ? -rel : rel;
  if (ap < 8) return ret + ap;
  // 8 + log(ap/8)/log(16) * 8 ; 8/ln(16) = 2.8853901
  int large = 8 + (int)(__logf((float)ap * 0.125f) * 2.8853901f);
  if (large > 15) large = 15;
  return ret + large;
}

// ---------------------------------------------------------------------------
// Flash attention, one (batch,head) x 128 query rows per block. 8 waves.
// QKV packed [B,S,3072] bf16 (Q|K|V each 1024 = 16 heads x 64).
// Out: [B,S,1024] bf16.  K block staged via TDM (tensor_load_to_lds) when
// available; V staged transposed by hand (TDM has no transpose).
// Softmax row-sums computed on the matrix pipe: rs = P x ones (2 WMMAs)
// instead of 32 ds_bpermute butterflies per key block.
// ---------------------------------------------------------------------------
__global__ __launch_bounds__(256) void t5_flash_attn(const __bf16* __restrict__ qkv,
                                                     const int* __restrict__ amask,
                                                     const float* __restrict__ rel_bias,
                                                     __bf16* __restrict__ out) {
  constexpr int S = 2048, HD = 64, NH = 16, ROW = 3072;
  constexpr int LDK = 72;                       // 144B rows: 16B aligned

  __shared__ __align__(16) __bf16 Ks[64 * LDK];        // K block, [key][d]
  __shared__ __align__(16) __bf16 Vt[64 * LDK];        // V block transposed, [d][key]
  __shared__ __align__(16) __bf16 Ps[8][16 * LDK];     // per-wave P tile
  __shared__ float lut[256];                           // bias LUT over clamp(j-i)

  const int qbase = blockIdx.x * 128;
  const int b     = blockIdx.y >> 4;
  const int h     = blockIdx.y & 15;
  const int t     = threadIdx.x;
  const int wave  = t >> 5;
  const int lane  = t & 31;
  const int colL  = lane & 15;
  const int rowHi = (lane >> 4) * 8;

  // Positional-bias LUT: bias(rel) for rel in [-128, 127] (exact: bucket
  // saturates for |rel| >= 91).  256 bucket evals per block total.
  lut[t] = rel_bias[rel_bucket(t - 128) * NH + h];

  // Q fragments for this wave's 16 rows, live in registers for the whole kernel.
  const __bf16* Qb = qkv + ((size_t)(b * S + qbase + wave * 16)) * ROW + h * HD;
  const v16bf qa0 = load_frag_a(Qb, ROW);        // d 0..31
  const v16bf qa1 = load_frag_a(Qb + 32, ROW);   // d 32..63
  const v16bf onesb = ones_frag();

  v8f   o[4] = {};
  float mrow[8], lrow[8];
  #pragma unroll
  for (int r = 0; r < 8; ++r) { mrow[r] = -3.0e38f; lrow[r] = 0.f; }

  const int qrow0 = qbase + wave * 16 + rowHi;   // this lane's first C-row

  for (int kb = 0; kb < S; kb += 64) {
    // ---- stage K block [64 keys x 64 d], row-major, LDS stride 72 ----
#if HAVE_TDM
    if (t == 0) {
      const __bf16* Kg = qkv + ((size_t)(b * S + kb)) * ROW + 1024 + h * HD;
      tdm_load_2d_bf16((unsigned)(unsigned long long)(uintptr_t)(void*)Ks,
                       (unsigned long long)(uintptr_t)Kg,
                       /*tensor_d0=*/64, /*tensor_d1=*/64,
                       /*d0_stride=*/ROW,
                       /*tile_d0=*/64, /*tile_d1=*/64);
    }
#else
    for (int i = t; i < 512; i += 256) {
      const int s = i >> 3, c = (i & 7) * 8;
      const u32x4* g = (const u32x4*)(qkv + ((size_t)(b * S + kb + s)) * ROW
                                      + 1024 + h * HD + c);
      *(u32x4*)(Ks + s * LDK + c) = *g;
    }
#endif
    // ---- stage V block transposed : 1024 chunks of 4 bf16 ----
    for (int i = t; i < 1024; i += 256) {
      const int s = i >> 4, c = (i & 15) * 4;
      union { u32x2 u; __bf16 e[4]; } tmp;
      tmp.u = *(const u32x2*)(qkv + ((size_t)(b * S + kb + s)) * ROW
                              + 2048 + h * HD + c);
      #pragma unroll
      for (int j = 0; j < 4; ++j) Vt[(c + j) * LDK + s] = tmp.e[j];
    }
#if HAVE_TDM
    if (t == 0) __builtin_amdgcn_s_wait_tensorcnt(0);   // TDM completion
#endif
    __syncthreads();

    // ---- scores: 16x64 = 4 n-tiles, each accumulated over d in 2 WMMAs ----
    v8f sc[4];
    #pragma unroll
    for (int n = 0; n < 4; ++n) {
      v8f a = {};
      a = wmma_bf16(qa0, load_frag_b_nt(Ks + (n * 16) * LDK, LDK), a);
      a = wmma_bf16(qa1, load_frag_b_nt(Ks + (n * 16) * LDK + 32, LDK), a);
      sc[n] = a;
    }

    // ---- T5 positional bias + mask via LDS LUT ----
    #pragma unroll
    for (int n = 0; n < 4; ++n) {
      const int j = kb + n * 16 + colL;
      const bool masked = (amask[(size_t)b * S + j] == 0);
      #pragma unroll
      for (int r = 0; r < 8; ++r) {
        int rel = j - (qrow0 + r);
        rel = (rel < -128) ? -128 : (rel > 127 ? 127 : rel);
        const float bias = masked ? -3.4e38f : lut[rel + 128];
        sc[n][r] += bias;
      }
    }

    // ---- softmax part 1: running max (butterfly), exponentiate, rescale ----
    #pragma unroll
    for (int r = 0; r < 8; ++r) {
      float vmax = fmaxf(fmaxf(sc[0][r], sc[1][r]), fmaxf(sc[2][r], sc[3][r]));
      #pragma unroll
      for (int off = 1; off < 16; off <<= 1)
        vmax = fmaxf(vmax, __shfl_xor(vmax, off, 32));
      const float mnew  = fmaxf(mrow[r], vmax);
      const float scale = __expf(mrow[r] - mnew);
      mrow[r] = mnew;
      #pragma unroll
      for (int n = 0; n < 4; ++n) sc[n][r] = __expf(sc[n][r] - mnew);
      lrow[r] *= scale;
      #pragma unroll
      for (int dt = 0; dt < 4; ++dt) o[dt][r] *= scale;
    }

    // ---- C-layout P -> A-layout via per-wave LDS round trip ----
    #pragma unroll
    for (int n = 0; n < 4; ++n)
      #pragma unroll
      for (int r = 0; r < 8; ++r)
        Ps[wave][(rowHi + r) * LDK + n * 16 + colL] = (__bf16)sc[n][r];
    // same-wave LDS store->load ordering handled via DScnt waits (compiler)

    const v16bf pa0 = load_frag_a(&Ps[wave][0], LDK);        // keys 0..31
    const v16bf pa1 = load_frag_a(&Ps[wave][0] + 32, LDK);   // keys 32..63

    // ---- softmax part 2: row-sums on the matrix pipe (P x ones) ----
    // D[r][*] = sum_j P[r][j], broadcast across all 16 lanes in C layout.
    {
      v8f rs = {};
      rs = wmma_bf16(pa0, onesb, rs);
      rs = wmma_bf16(pa1, onesb, rs);
      #pragma unroll
      for (int r = 0; r < 8; ++r) lrow[r] += rs[r];
    }

    // ---- O += P * V : contraction over 64 keys in 2 WMMA k-steps ----
    #pragma unroll
    for (int dt = 0; dt < 4; ++dt) {
      o[dt] = wmma_bf16(pa0, load_frag_b_nt(Vt + (dt * 16) * LDK, LDK), o[dt]);
      o[dt] = wmma_bf16(pa1, load_frag_b_nt(Vt + (dt * 16) * LDK + 32, LDK), o[dt]);
    }
    __syncthreads();   // before next iteration overwrites Ks / Vt
  }

  // ---- normalize and store [B,S,1024] bf16 ----
  float inv[8];
  #pragma unroll
  for (int r = 0; r < 8; ++r) inv[r] = 1.0f / lrow[r];
  __bf16* ob = out + ((size_t)(b * S + qbase + wave * 16)) * 1024 + h * HD;
  #pragma unroll
  for (int dt = 0; dt < 4; ++dt)
    #pragma unroll
    for (int r = 0; r < 8; ++r)
      ob[(size_t)(rowHi + r) * 1024 + dt * 16 + colL] = (__bf16)(o[dt][r] * inv[r]);
}

// ---------------------------------------------------------------------------
// Host driver
// ---------------------------------------------------------------------------
extern "C" void kernel_launch(void* const* d_in, const int* in_sizes, int n_in,
                              void* d_out, int out_size, void* d_ws, size_t ws_size,
                              hipStream_t stream) {
  (void)in_sizes; (void)n_in; (void)out_size; (void)ws_size;

  constexpr int B = 2, S = 2048, D = 1024, NQKV = 3072;
  constexpr size_t MS = (size_t)B * S;          // 4096 rows

  const float* hidden   = (const float*)d_in[0];
  const int*   amask    = (const int*)d_in[1];
  const float* w_qkv    = (const float*)d_in[2];
  const float* rel_bias = (const float*)d_in[3];
  const float* w_o      = (const float*)d_in[4];
  float*       out      = (float*)d_out;

  __bf16* Xb    = (__bf16*)d_ws;                      // [4096,1024]
  __bf16* Wqkvb = Xb    + MS * D;                     // [3072,1024]
  __bf16* Wob   = Wqkvb + (size_t)NQKV * D;           // [1024,1024]
  __bf16* QKVb  = Wob   + (size_t)D * D;              // [4096,3072]
  __bf16* AOb   = QKVb  + MS * NQKV;                  // [4096,1024]

  // fp32 -> bf16 conversions
  {
    int n4 = (int)(MS * D / 4);
    cvt_f32_bf16_x4<<<(n4 + 255) / 256, 256, 0, stream>>>((const float4*)hidden, (u32x2*)Xb, n4);
    n4 = NQKV * D / 4;
    cvt_f32_bf16_x4<<<(n4 + 255) / 256, 256, 0, stream>>>((const float4*)w_qkv, (u32x2*)Wqkvb, n4);
    n4 = D * D / 4;
    cvt_f32_bf16_x4<<<(n4 + 255) / 256, 256, 0, stream>>>((const float4*)w_o, (u32x2*)Wob, n4);
  }

  // QKV = X @ Wqkv^T   (bf16 out)
  gemm_nt_bf16<__bf16><<<dim3(NQKV / 64, MS / 128), 256, 0, stream>>>(
      Xb, Wqkvb, QKVb, (int)MS, NQKV, D);

  // flash attention per (b,h), 128 query rows / block, 8 waves
  t5_flash_attn<<<dim3(S / 128, B * 16), 256, 0, stream>>>(QKVb, amask, rel_bias, AOb);

  // out = AO @ Wo^T    (fp32 out)
  gemm_nt_bf16<float><<<dim3(D / 64, MS / 128), 256, 0, stream>>>(
      AOb, Wob, out, (int)MS, D, D);
}
